// KMultiheadAttention_6356551598117
// MI455X (gfx1250) — matrix operand order repaired and checked by reading, not compiled, not run on hardware
//
#include <hip/hip_runtime.h>
#include <hip/hip_bf16.h>
#include <math.h>

// Problem constants (match reference)
#define BB 4
#define LL 2048
#define DD 1024
// GEMM tiling: 256 threads = 8 waves (wave32), waves arranged 4(M) x 2(N);
// each wave owns a 32x64 tile = 2x4 WMMA 16x16 tiles (8 v_wmma per K-step).
#define BM 128
#define BN 128
#define BK 32
#define LDSK 40   // BK + 8 bf16 pad -> row stride 80 bytes (16B aligned)

typedef __attribute__((ext_vector_type(16))) __bf16 v16bf;
typedef __attribute__((ext_vector_type(8)))  float  v8f;

union Frag { v16bf v; uint4 u[2]; };

// MODE 0: C = scale*(A@B[T]) + bias      (bias may be null)
// MODE 1: C = masked(scale*(A@Bᵀ))       (attention scores with NEG mask fill)
// TRANSB 1: B memory is [N, K] (nn.Linear weight, or kᵀ for scores)
// TRANSB 0: B memory is [K, N] (v matrix for attn@v)
template<int MODE, int TRANSB>
__global__ __launch_bounds__(256)
void gemm_bf16_wmma(const float* __restrict__ A, int lda,
                    const float* __restrict__ Bmat, int ldb,
                    float* __restrict__ C, int ldc,
                    const float* __restrict__ bias,
                    float scale, int K,
                    long long sA, long long sB, long long sC,
                    const unsigned char* __restrict__ attn_mask,
                    const unsigned char* __restrict__ kpm)
{
    const int bz = blockIdx.z;
    A    += (long long)bz * sA;
    Bmat += (long long)bz * sB;
    C    += (long long)bz * sC;

    const int mTile = blockIdx.y * BM;
    const int nTile = blockIdx.x * BN;

    __shared__ alignas(16) __bf16 As[BM * LDSK];
    __shared__ alignas(16) __bf16 Bs[BN * LDSK];

    const int tid   = threadIdx.x;
    const int lane  = tid & 31;
    const int wave  = tid >> 5;
    const int wm    = (wave & 3) * 32;   // 4 waves along M
    const int wn    = (wave >> 2) * 64;  // 2 waves along N
    const int l16   = lane & 15;
    const int lhalf = lane >> 4;         // 0: lanes 0-15, 1: lanes 16-31

    v8f acc[2][4] = {};

    // A-tile fill mapping: 2 threads per row, 16 consecutive k each
    const int ar  = tid >> 1;            // 0..127
    const int akh = (tid & 1) * 16;      // 0 or 16
    // B-tile fill mappings
    const int bnT = tid >> 1;            // TRANSB: 0..127 rows of B[N,K]
    const int bkT = (tid & 1) * 16;
    const int bkN = tid >> 3;            // !TRANSB: 0..31 rows of B[K,N]
    const int bnN = (tid & 7) * 16;      // 16 consecutive n per thread

    for (int k0 = 0; k0 < K; k0 += BK) {
        // ---- stage A tile (BM x BK), fp32 -> bf16 ----
        {
            const float4* ga4 = (const float4*)(A + (size_t)(mTile + ar) * lda + k0 + akh);
            __bf16* la = &As[ar * LDSK + akh];
            #pragma unroll
            for (int j = 0; j < 4; ++j) {
                float4 f = ga4[j];
                la[4*j+0] = (__bf16)f.x; la[4*j+1] = (__bf16)f.y;
                la[4*j+2] = (__bf16)f.z; la[4*j+3] = (__bf16)f.w;
            }
        }
        // ---- stage B tile (BN x BK) as [n][k] ----
        if (TRANSB) {
            const float4* gb4 = (const float4*)(Bmat + (size_t)(nTile + bnT) * ldb + k0 + bkT);
            __bf16* lb = &Bs[bnT * LDSK + bkT];
            #pragma unroll
            for (int j = 0; j < 4; ++j) {
                float4 f = gb4[j];
                lb[4*j+0] = (__bf16)f.x; lb[4*j+1] = (__bf16)f.y;
                lb[4*j+2] = (__bf16)f.z; lb[4*j+3] = (__bf16)f.w;
            }
        } else {
            const float4* gb4 = (const float4*)(Bmat + (size_t)(k0 + bkN) * ldb + nTile + bnN);
            #pragma unroll
            for (int j = 0; j < 4; ++j) {
                float4 f = gb4[j];
                Bs[(bnN + 4*j + 0) * LDSK + bkN] = (__bf16)f.x;
                Bs[(bnN + 4*j + 1) * LDSK + bkN] = (__bf16)f.y;
                Bs[(bnN + 4*j + 2) * LDSK + bkN] = (__bf16)f.z;
                Bs[(bnN + 4*j + 3) * LDSK + bkN] = (__bf16)f.w;
            }
        }
        // Prefetch next K-tile (lowers to global_prefetch_b8; no LOADcnt cost)
        if (k0 + BK < K) {
            __builtin_prefetch(A + (size_t)(mTile + ar) * lda + k0 + BK + akh, 0, 1);
            if (TRANSB)
                __builtin_prefetch(Bmat + (size_t)(nTile + bnT) * ldb + k0 + BK + bkT, 0, 1);
            else
                __builtin_prefetch(Bmat + (size_t)(k0 + BK + bkN) * ldb + nTile + bnN, 0, 1);
        }
        __syncthreads();

        // ---- build fragments per ISA 16-bit layouts (two ds_load_b128 each) ----
        Frag afr[2], bfr[4];
        const int aoff = lhalf ? 8 : 0;   // A: lanes<16 -> K {0..7,16..23}; >=16 -> {8..15,24..31}
        #pragma unroll
        for (int mi = 0; mi < 2; ++mi) {
            const __bf16* p = &As[(wm + mi*16 + l16) * LDSK + aoff];
            afr[mi].u[0] = *(const uint4*)(p);
            afr[mi].u[1] = *(const uint4*)(p + 16);
        }
        const int boff = lhalf ? 8 : 0;   // B: lanes<16 -> K {0..7,16..23}; >=16 -> {8..15,24..31}
        #pragma unroll
        for (int ni = 0; ni < 4; ++ni) {
            const __bf16* p = &Bs[(wn + ni*16 + l16) * LDSK + boff];
            bfr[ni].u[0] = *(const uint4*)(p);
            bfr[ni].u[1] = *(const uint4*)(p + 16);
        }
        #pragma unroll
        for (int mi = 0; mi < 2; ++mi)
            #pragma unroll
            for (int ni = 0; ni < 4; ++ni)
                acc[mi][ni] = __builtin_amdgcn_wmma_f32_16x16x32_bf16(
                    false, afr[mi].v, false, bfr[ni].v,
                    (short)0, acc[mi][ni], false, false);
        __syncthreads();
    }

    // ---- epilogue: C/D layout -> lane l16 = N, vgpr v (+8 for lanes>=16) = M ----
    #pragma unroll
    for (int ni = 0; ni < 4; ++ni) {
        const int ng = nTile + wn + ni*16 + l16;
        float bval = 0.0f;
        if (MODE == 0) { if (bias) bval = bias[ng]; }
        bool kmask = false;
        if (MODE == 1) kmask = (kpm[bz * LL + ng] != 0);
        #pragma unroll
        for (int mi = 0; mi < 2; ++mi) {
            #pragma unroll
            for (int v = 0; v < 8; ++v) {
                const int mg = mTile + wm + mi*16 + lhalf*8 + v;
                float r = acc[mi][ni][v] * scale + bval;
                if (MODE == 1) {
                    if (kmask || attn_mask[(size_t)mg * LL + ng]) r = -1e9f;
                }
                C[(size_t)mg * ldc + ng] = r;
            }
        }
    }
}

// Row softmax over L=2048, one 256-thread block (8 wave32) per row, values kept in regs
__global__ __launch_bounds__(256)
void softmax_rows(float* __restrict__ attn)
{
    const int row = blockIdx.x;
    float* p = attn + (size_t)row * LL;
    const int tid = threadIdx.x;
    __shared__ float redmax[8];
    __shared__ float redsum[8];

    float vals[8];
    float mx = -3.0e38f;
    #pragma unroll
    for (int j = 0; j < 8; ++j) { vals[j] = p[tid + j*256]; mx = fmaxf(mx, vals[j]); }
    #pragma unroll
    for (int o = 16; o > 0; o >>= 1) mx = fmaxf(mx, __shfl_xor(mx, o, 32));
    if ((tid & 31) == 0) redmax[tid >> 5] = mx;
    __syncthreads();
    mx = redmax[0];
    #pragma unroll
    for (int j = 1; j < 8; ++j) mx = fmaxf(mx, redmax[j]);

    float sum = 0.0f;
    #pragma unroll
    for (int j = 0; j < 8; ++j) { vals[j] = __expf(vals[j] - mx); sum += vals[j]; }
    #pragma unroll
    for (int o = 16; o > 0; o >>= 1) sum += __shfl_xor(sum, o, 32);
    if ((tid & 31) == 0) redsum[tid >> 5] = sum;
    __syncthreads();
    sum = 0.0f;
    #pragma unroll
    for (int j = 0; j < 8; ++j) sum += redsum[j];
    const float inv = 1.0f / sum;
    #pragma unroll
    for (int j = 0; j < 8; ++j) p[tid + j*256] = vals[j] * inv;
}

extern "C" void kernel_launch(void* const* d_in, const int* in_sizes, int n_in,
                              void* d_out, int out_size, void* d_ws, size_t ws_size,
                              hipStream_t stream)
{
    const float* query = (const float*)d_in[0];
    const float* key   = (const float*)d_in[1];
    const float* value = (const float*)d_in[2];
    const unsigned char* kpm   = (const unsigned char*)d_in[3];  // bool (1 byte)
    const unsigned char* amask = (const unsigned char*)d_in[4];  // bool (1 byte)
    const float* Wq = (const float*)d_in[5];
    const float* bq = (const float*)d_in[6];
    const float* Wk = (const float*)d_in[7];
    const float* bk = (const float*)d_in[8];
    const float* Wv = (const float*)d_in[9];
    const float* bv = (const float*)d_in[10];
    const float* Wo = (const float*)d_in[11];
    const float* bo = (const float*)d_in[12];

    const long long MD = (long long)BB * LL * DD;   // 8,388,608
    float* q   = (float*)d_ws;      // fp32 intermediates in workspace
    float* k   = q + MD;
    float* v   = k + MD;
    float* ctx = v + MD;

    float* out  = (float*)d_out;            // (B, L, D)
    float* attn = out + MD;                 // (B, 1, L, L)

    dim3 blk(256);
    dim3 gproj(DD / BN, (BB * LL) / BM, 1); // 8 x 64

    // Q/K/V projections: y = x @ Wᵀ + b
    gemm_bf16_wmma<0,1><<<gproj, blk, 0, stream>>>(query, DD, Wq, DD, q, DD, bq,
        1.0f, DD, 0, 0, 0, nullptr, nullptr);
    gemm_bf16_wmma<0,1><<<gproj, blk, 0, stream>>>(key,   DD, Wk, DD, k, DD, bk,
        1.0f, DD, 0, 0, 0, nullptr, nullptr);
    gemm_bf16_wmma<0,1><<<gproj, blk, 0, stream>>>(value, DD, Wv, DD, v, DD, bv,
        1.0f, DD, 0, 0, 0, nullptr, nullptr);

    // Scores: S = q @ kᵀ / sqrt(D), masks applied in epilogue -> attn region of d_out
    dim3 gsc(LL / BN, LL / BM, BB);         // 16 x 16 x 4
    gemm_bf16_wmma<1,1><<<gsc, blk, 0, stream>>>(q, DD, k, DD, attn, LL, nullptr,
        1.0f / 32.0f, DD,
        (long long)LL * DD, (long long)LL * DD, (long long)LL * LL, amask, kpm);

    // Row softmax in place (this IS the attn output)
    softmax_rows<<<dim3(BB * LL), blk, 0, stream>>>(attn);

    // Context: ctx = attn @ v
    dim3 gcx(DD / BN, LL / BM, BB);         // 8 x 16 x 4
    gemm_bf16_wmma<0,0><<<gcx, blk, 0, stream>>>(attn, LL, v, DD, ctx, DD, nullptr,
        1.0f, LL,
        (long long)LL * LL, (long long)LL * DD, (long long)LL * DD, nullptr, nullptr);

    // Output projection: out = ctx @ Woᵀ + bo
    gemm_bf16_wmma<0,1><<<gproj, blk, 0, stream>>>(ctx, DD, Wo, DD, out, DD, bo,
        1.0f, DD, 0, 0, 0, nullptr, nullptr);
}
